// CrissCrossAttention_48954037240512
// MI455X (gfx1250) — matrix operand order repaired and checked by reading, not compiled, not run on hardware
//
#include <hip/hip_runtime.h>
#include <hip/hip_bf16.h>

typedef __attribute__((ext_vector_type(16))) _Float16 v16h;
typedef __attribute__((ext_vector_type(8)))  _Float16 v8h;
typedef __attribute__((ext_vector_type(8)))  float    v8f;
typedef __attribute__((ext_vector_type(4))) unsigned int u32x4;
typedef __attribute__((ext_vector_type(4))) int  i32x4;
typedef __attribute__((ext_vector_type(8))) int  i32x8;

#define B_  4
#define C_  256
#define CB_ 32
#define N_  4096

// ---------------------------------------------------------------------------
// Projections: q,k stored transposed [B, N, 32] f16 (rows contiguous for WMMA
// A/B operands); v stored [B, C, N] f16 (rows contiguous in key index m).
// ---------------------------------------------------------------------------
__global__ __launch_bounds__(256)
void proj_qk(const float* __restrict__ x,
             const float* __restrict__ Wq, const float* __restrict__ bq,
             const float* __restrict__ Wk, const float* __restrict__ bk,
             _Float16* __restrict__ qT, _Float16* __restrict__ kT)
{
    int gid = blockIdx.x * blockDim.x + threadIdx.x;   // B*N threads
    int b = gid / N_;
    int n = gid - b * N_;
    float qa[CB_], ka[CB_];
    #pragma unroll
    for (int d = 0; d < CB_; ++d) { qa[d] = bq[d]; ka[d] = bk[d]; }
    const float* xp = x + (size_t)b * C_ * N_ + n;
    for (int c = 0; c < C_; ++c) {
        float xv = xp[(size_t)c * N_];
        #pragma unroll
        for (int d = 0; d < CB_; ++d) {
            qa[d] = fmaf(Wq[d * C_ + c], xv, qa[d]);
            ka[d] = fmaf(Wk[d * C_ + c], xv, ka[d]);
        }
    }
    _Float16* qrow = qT + (size_t)gid * CB_;
    _Float16* krow = kT + (size_t)gid * CB_;
    #pragma unroll
    for (int d = 0; d < CB_; ++d) {
        qrow[d] = (_Float16)qa[d];
        krow[d] = (_Float16)ka[d];
    }
}

__global__ __launch_bounds__(256)
void proj_v(const float* __restrict__ x, const float* __restrict__ Wv,
            const float* __restrict__ bv, _Float16* __restrict__ vF)
{
    const int nt = N_ / 256;
    int bc    = blockIdx.x / nt;
    int ntile = blockIdx.x - bc * nt;
    int b = bc / C_;
    int c = bc - b * C_;
    int n = ntile * 256 + threadIdx.x;
    const float* xcol = x + (size_t)b * C_ * N_ + n;
    const float* w    = Wv + (size_t)c * C_;
    float acc = bv[c];
    #pragma unroll 8
    for (int cc = 0; cc < C_; ++cc)
        acc = fmaf(w[cc], xcol[(size_t)cc * N_], acc);
    vF[((size_t)(b * C_ + c)) * N_ + n] = (_Float16)acc;
}

// ---------------------------------------------------------------------------
// Operand loaders per CDNA5 WMMA VGPR layouts (ISA 7.12.2), wave32.
// A (16x32 f16): lane L and L+16 hold row M = L&15; K halves {0..7,16..23}
//                for lanes<16, {8..15,24..31} for lanes>=16.
// B (32x16 f16): lane holds col N = lane&15; same K halves as A
//                ({0..7,16..23} for lanes<16, {8..15,24..31} for lanes>=16).
// ---------------------------------------------------------------------------
__device__ inline v16h load_A_row32(const _Float16* row_base, int half)
{
    const v8h* p = (const v8h*)(row_base + half * 8);
    v8h lo = p[0];      // K = half*8 .. half*8+7
    v8h hi = p[2];      // K = half*8+16 .. half*8+23
    v16h a;
    #pragma unroll
    for (int t = 0; t < 8; ++t) { a[t] = lo[t]; a[8 + t] = hi[t]; }
    return a;
}

__device__ inline v16h load_B_col32(const _Float16* row_base, int half)
{
    const v8h* p = (const v8h*)(row_base + half * 8);
    v8h lo = p[0], hi = p[2];   // K = half*8..+7 and half*8+16..+23
    v16h bb;
    #pragma unroll
    for (int t = 0; t < 8; ++t) { bb[t] = lo[t]; bb[8 + t] = hi[t]; }
    return bb;
}

// ---------------------------------------------------------------------------
// Phase A: per-query-row softmax stats  M_m = max_n S[m,n],  L_m = sum exp.
// ---------------------------------------------------------------------------
__global__ __launch_bounds__(256)
void attn_stats(const _Float16* __restrict__ qT, const _Float16* __restrict__ kT,
                float* __restrict__ Mst, float* __restrict__ Lst)
{
    const int lane = threadIdx.x & 31;
    const int wave = threadIdx.x >> 5;
    const int gw   = blockIdx.x * 8 + wave;   // 0..1023
    const int b    = gw >> 8;
    const int m0   = (gw & 255) << 4;
    const int half = lane >> 4;
    const int l16  = lane & 15;

    v16h qa = load_A_row32(qT + ((size_t)(b * N_ + m0 + l16)) * CB_, half);

    float rmax[8], rsum[8];
    #pragma unroll
    for (int r = 0; r < 8; ++r) { rmax[r] = -__builtin_inff(); rsum[r] = 0.0f; }

    const _Float16* kb0 = kT + (size_t)b * N_ * CB_;
    for (int n0 = 0; n0 < N_; n0 += 16) {
        v16h kb = load_B_col32(kb0 + (size_t)(n0 + l16) * CB_, half);
        v8f z = {};
        v8f s = __builtin_amdgcn_wmma_f32_16x16x32_f16(
                    false, qa, false, kb, (short)0, z, false, false);
        #pragma unroll
        for (int r = 0; r < 8; ++r) {
            float v  = s[r];
            float mn = fmaxf(rmax[r], v);
            rsum[r]  = rsum[r] * __expf(rmax[r] - mn) + __expf(v - mn);
            rmax[r]  = mn;
        }
    }
    // merge the 16 per-lane partials within each half-wave
    #pragma unroll
    for (int r = 0; r < 8; ++r) {
        #pragma unroll
        for (int off = 1; off < 16; off <<= 1) {
            float om = __shfl_xor(rmax[r], off, 32);
            float os = __shfl_xor(rsum[r], off, 32);
            float mn = fmaxf(rmax[r], om);
            rsum[r]  = rsum[r] * __expf(rmax[r] - mn) + os * __expf(om - mn);
            rmax[r]  = mn;
        }
    }
    if (l16 == 0) {
        #pragma unroll
        for (int r = 0; r < 8; ++r) {
            int m = m0 + half * 8 + r;
            Mst[b * N_ + m] = rmax[r];
            Lst[b * N_ + m] = rsum[r];
        }
    }
}

// ---------------------------------------------------------------------------
// Phase B: out[c, n] = sum_m v[c,m] * exp(S[m,n]-M_m)/L_m ; +x residual.
// All 8 waves of a block share batch b and the same V tiles, so one wave
// DMAs V[0..255][m0..m0+31] (16 KB) into LDS with the Tensor Data Mover
// (tensor_load_to_lds, TENSORcnt), and all waves feed the PV WMMAs from LDS.
// ---------------------------------------------------------------------------
__global__ __launch_bounds__(256)
void attn_out(const float* __restrict__ x,
              const _Float16* __restrict__ qT, const _Float16* __restrict__ kT,
              const _Float16* __restrict__ vF,
              const float* __restrict__ Mst, const float* __restrict__ Lst,
              float* __restrict__ out)
{
    __shared__ _Float16 vtile[C_ * 32];    // 16 KB: V[c][m-local], 64B rows

    const int lane = threadIdx.x & 31;
    const int wave = threadIdx.x >> 5;
    const int gw   = blockIdx.x * 8 + wave;   // 0..1023
    const int b    = gw >> 8;                 // 32 blocks per batch -> uniform b
    const int n0   = (gw & 255) << 4;
    const int half = lane >> 4;
    const int l16  = lane & 15;

    // fixed B operand: key columns n0..n0+15
    v16h kb = load_B_col32(kT + ((size_t)(b * N_ + n0 + l16)) * CB_, half);

    v8f acc[16];
    #pragma unroll
    for (int i = 0; i < 16; ++i) { v8f z = {}; acc[i] = z; }

    const _Float16* qb0 = qT + (size_t)b * N_ * CB_;
    const _Float16* vb0 = vF + (size_t)b * C_ * N_;
    const float*    Mb  = Mst + b * N_;
    const float*    Lb  = Lst + b * N_;

    const unsigned long long vbase = (unsigned long long)(uintptr_t)vb0;
    const unsigned int lds_addr    = (unsigned int)(uintptr_t)(&vtile[0]);

    for (int m0 = 0; m0 < N_; m0 += 32) {
        __syncthreads();               // previous tile fully consumed
        if (wave == 0) {
            // --- TDM descriptor (ISA 8.3/8.4): 2D tile, 32 x 256, f16 ---
            unsigned long long ga = vbase + (unsigned long long)m0 * 2ull;
            u32x4 g0;
            g0[0] = 1u;                                   // count=1 (valid D#)
            g0[1] = lds_addr;                             // lds_addr (bytes)
            g0[2] = (unsigned int)ga;                     // global_addr[31:0]
            g0[3] = (unsigned int)((ga >> 32) & 0x1FFFFFFull) | (2u << 30); // [56:32] | type=2
            i32x8 g1;
            g1[0] = 0x00010000;                // data_size=2B; no mask/pad/iter
            g1[1] = (int)(N_ << 16);           // tensor_dim0 = 4096
            g1[2] = (int)(C_ << 16);           // tensor_dim1 = 256
            g1[3] = (int)(32 << 16);           // tile_dim0 = 32
            g1[4] = 256;                       // tile_dim1 = 256, tile_dim2 = 0
            g1[5] = N_;                        // tensor_dim0_stride = 4096
            g1[6] = 0;
            g1[7] = 0;
            i32x4 z4 = {};
#if __clang_major__ >= 23
            i32x8 z8 = {};
            __builtin_amdgcn_tensor_load_to_lds(g0, g1, z4, z4, z8, 0);
#else
            __builtin_amdgcn_tensor_load_to_lds(g0, g1, z4, z4, 0);
#endif
            __builtin_amdgcn_s_wait_tensorcnt(0);
        }
        __syncthreads();               // tile visible to all waves

        v16h qa0 = load_A_row32(qb0 + (size_t)(m0 + l16) * CB_, half);
        v16h qa1 = load_A_row32(qb0 + (size_t)(m0 + 16 + l16) * CB_, half);
        v8f z = {};
        v8f st0 = __builtin_amdgcn_wmma_f32_16x16x32_f16(
                      false, qa0, false, kb, (short)0, z, false, false);
        v8f st1 = __builtin_amdgcn_wmma_f32_16x16x32_f16(
                      false, qa1, false, kb, (short)0, z, false, false);

        // normalize: VGPR r of st0 holds row m = m0 + half*8 + r, st1 is +16
        const float* Mp = Mb + m0 + half * 8;
        const float* Lp = Lb + m0 + half * 8;
        v8f p0, p1;
        #pragma unroll
        for (int r = 0; r < 8; ++r) {
            p0[r] = __fdividef(__expf(st0[r] - Mp[r]),      Lp[r]);
            p1[r] = __fdividef(__expf(st1[r] - Mp[16 + r]), Lp[16 + r]);
        }

        // D-layout (rows=m over VGPRs, cols=n over lanes) -> B-operand
        // (K=m pairs per VGPR, cols=n) with one half-wave shfl_xor(16).
        v16h pb;
        #pragma unroll
        for (int j = 0; j < 8; ++j) {
            const int r = (j < 4) ? 2 * j : 2 * j - 8;
            float s0lo = p0[r],     s0hi = p0[r + 1];
            float s1lo = p1[r],     s1hi = p1[r + 1];
            float x0lo = __shfl_xor(s0lo, 16, 32);
            float x0hi = __shfl_xor(s0hi, 16, 32);
            float x1lo = __shfl_xor(s1lo, 16, 32);
            float x1hi = __shfl_xor(s1hi, 16, 32);
            float blo, bhi;
            if (j < 4) { blo = half ? x1lo : s0lo;  bhi = half ? x1hi : s0hi; }
            else       { blo = half ? s1lo : x0lo;  bhi = half ? s1hi : x0hi; }
            pb[2 * j]     = (_Float16)blo;
            pb[2 * j + 1] = (_Float16)bhi;
        }

        // acc[i] += V(cblock i, m0..m0+31) x P, A-operand fed from LDS tile
        #pragma unroll
        for (int i = 0; i < 16; ++i) {
            const _Float16* vr = vtile + (size_t)(i * 16 + l16) * 32;
            const v8h* p = (const v8h*)(vr + half * 16);
            v8h lo = p[0], hi = p[1];          // m {half*16..+15}
            v16h va;
            #pragma unroll
            for (int t = 0; t < 8; ++t) { va[t] = lo[t]; va[8 + t] = hi[t]; }
            acc[i] = __builtin_amdgcn_wmma_f32_16x16x32_f16(
                         false, va, false, pb, (short)0, acc[i], false, false);
        }
    }

    // Epilogue: D layout -> out[b, c, n] with residual
    #pragma unroll
    for (int i = 0; i < 16; ++i) {
        #pragma unroll
        for (int r = 0; r < 8; ++r) {
            int c = i * 16 + half * 8 + r;
            size_t idx = ((size_t)(b * C_ + c)) * N_ + (n0 + l16);
            out[idx] = acc[i][r] + x[idx];
        }
    }
}

// ---------------------------------------------------------------------------
extern "C" void kernel_launch(void* const* d_in, const int* in_sizes, int n_in,
                              void* d_out, int out_size, void* d_ws, size_t ws_size,
                              hipStream_t stream)
{
    (void)in_sizes; (void)n_in; (void)out_size; (void)ws_size;
    const float* x  = (const float*)d_in[0];
    const float* Wq = (const float*)d_in[1];
    const float* bq = (const float*)d_in[2];
    const float* Wk = (const float*)d_in[3];
    const float* bk = (const float*)d_in[4];
    const float* Wv = (const float*)d_in[5];
    const float* bv = (const float*)d_in[6];
    float* out = (float*)d_out;

    _Float16* qT = (_Float16*)d_ws;                        // [B,N,32]
    _Float16* kT = qT + (size_t)B_ * N_ * CB_;             // [B,N,32]
    _Float16* vF = kT + (size_t)B_ * N_ * CB_;             // [B,C,N]
    float*   Mst = (float*)(vF + (size_t)B_ * C_ * N_);    // [B,N]
    float*   Lst = Mst + (size_t)B_ * N_;                  // [B,N]

    proj_qk  <<<(B_ * N_) / 256,        256, 0, stream>>>(x, Wq, bq, Wk, bk, qT, kT);
    proj_v   <<<B_ * C_ * (N_ / 256),   256, 0, stream>>>(x, Wv, bv, vF);
    attn_stats<<<(B_ * (N_ / 16)) / 8,  256, 0, stream>>>(qT, kT, Mst, Lst);
    attn_out <<<(B_ * (N_ / 16)) / 8,   256, 0, stream>>>(x, qT, kT, vF, Mst, Lst, out);
}
